// Graph_Generator_42571715838195
// MI455X (gfx1250) — matrix operand order repaired and checked by reading, not compiled, not run on hardware
//
#include <hip/hip_runtime.h>
#include <hip/hip_bf16.h>
#include <math.h>

#define BATCH 128
#define CH    128
#define NN    170
#define TT    12
#define NP    176   // nodes padded to 11*16
#define KP    192   // GEMM2 K padded to 6*32
#define NTIL  11    // 16-wide node tiles

typedef __attribute__((ext_vector_type(16))) __bf16 v16bf;
typedef __attribute__((ext_vector_type(2)))  __bf16 bf16x2;
typedef __attribute__((ext_vector_type(8)))  float  v8f;

// CDNA5 has a hardware tanh transcendental (v_tanh_f32); use it if exposed.
#if __has_builtin(__builtin_amdgcn_tanhf)
#define FAST_TANH(x) __builtin_amdgcn_tanhf(x)
#elif __has_builtin(__builtin_amdgcn_tanh_f32)
#define FAST_TANH(x) __builtin_amdgcn_tanh_f32(x)
#else
#define FAST_TANH(x) tanhf(x)
#endif

// A-matrix (16x32 bf16) per-lane K base for VGPR-pair j (ISA 7.12.2):
// lanes 0-15: j<4 -> K=2j, j>=4 -> K=16+2(j-4); lanes 16-31: +8
__device__ __forceinline__ int a_k(int j, int half) {
    return (j < 4 ? 2 * j : 16 + 2 * (j - 4)) + 8 * half;
}

// ------- Kernel 0: E_s f32 [C][N] -> bf16 TRANSPOSED [NP][CH] ---------------
__global__ void es_convert_kernel(const float* __restrict__ Es, __bf16* __restrict__ EsT) {
    int idx = blockIdx.x * 256 + threadIdx.x;   // idx over [NP][CH]
    if (idx >= NP * CH) return;
    int m = idx / CH, c = idx % CH;
    float v = (m < NN) ? Es[c * NN + m] : 0.f;
    EsT[idx] = (__bf16)v;
}

// ------- Kernel 1: t-reduce + GEMM1 (WMMA bf16) + tanh -> Ed[b][NP][KP] ------
__global__ __launch_bounds__(256) void gemm1_tanh_kernel(
        const float* __restrict__ x, const __bf16* __restrict__ EsT,
        __bf16* __restrict__ Ed) {
    __shared__ __bf16 xsb[16][132];  // even stride (4B-aligned pairs), 264B rows
    int b  = blockIdx.x / NTIL;
    int nt = blockIdx.x % NTIL;
    int n0 = nt * 16;
    int tid = threadIdx.x;

    // Phase 1: xsb[r][c] = bf16( sum_t x[b,c,n0+r,t] )  (single pass over x)
    for (int idx = tid; idx < 16 * CH; idx += 256) {
        int r = idx >> 7, c = idx & 127;
        int n = n0 + r;
        float s = 0.f;
        if (n < NN) {
            const float* p = x + (((size_t)b * CH + c) * NN + n) * TT;
#pragma unroll
            for (int t = 0; t < TT; ++t) s += p[t];
        }
        xsb[r][c] = (__bf16)s;
    }
    // Zero-fill Ed K-padding columns [NP, KP): 16 rows x 8 bf16-pairs as b32
    if (tid < 128) {
        int r = tid >> 3, c = NP + 2 * (tid & 7);
        *(uint*)&Ed[((size_t)b * NP + n0 + r) * KP + c] = 0u;
    }
    __syncthreads();

    int wave = tid >> 5, lane = tid & 31;
    int ln15 = lane & 15, half = lane >> 4;

    // Hoist A fragments (invariant across m-tiles): 4 k-steps of K=128
    v16bf afrag[4];
#pragma unroll
    for (int ks = 0; ks < 4; ++ks) {
#pragma unroll
        for (int j = 0; j < 8; ++j) {
            bf16x2 av = *(const bf16x2*)&xsb[ln15][a_k(j, half) + 32 * ks];
            afrag[ks][2 * j] = av.x; afrag[ks][2 * j + 1] = av.y;
        }
    }

    // Per wave: m-tiles of E_s; B pairs contiguous in EsT[m][c]
    for (int mt = wave; mt < NTIL; mt += 8) {
        int m0 = mt * 16;
        const __bf16* brow = EsT + (size_t)(m0 + ln15) * CH;
        v8f acc = {};
#pragma unroll
        for (int ks = 0; ks < 4; ++ks) {
            v16bf bb;
#pragma unroll
            for (int j = 0; j < 8; ++j) {
                bf16x2 bv = *(const bf16x2*)(brow + 32 * ks + a_k(j, half));
                bb[2 * j] = bv.x; bb[2 * j + 1] = bv.y;
            }
            acc = __builtin_amdgcn_wmma_f32_16x16x32_bf16(
                false, afrag[ks], false, bb, (short)0, acc, false, false);
        }
        // D layout: VGPR r -> M = r + 8*half, col = lane&15
#pragma unroll
        for (int r = 0; r < 8; ++r) {
            int M = r + 8 * half;
            float v = FAST_TANH(acc[r]);
            Ed[((size_t)b * NP + n0 + M) * KP + (m0 + ln15)] = (__bf16)v;
        }
    }
}

// -- Kernel 2: Gram rows (WMMA bf16) + scale/ReLU + softmax + mean-atomics ---
__global__ __launch_bounds__(256) void gemm2_softmax_kernel(
        const __bf16* __restrict__ Ed, float* __restrict__ accum) {
    __shared__ float sc[16][180];
    int b  = blockIdx.x / NTIL;
    int nt = blockIdx.x % NTIL;
    int n0 = nt * 16;
    int tid = threadIdx.x, wave = tid >> 5, lane = tid & 31;
    int ln15 = lane & 15, half = lane >> 4;
    const __bf16* EdB = Ed + (size_t)b * NP * KP;

    // Hoist A fragments (rows n0..n0+15, invariant across k-tiles): 6 k-steps
    v16bf afrag[6];
    {
        const __bf16* arow = EdB + (size_t)(n0 + ln15) * KP;
#pragma unroll
        for (int ks = 0; ks < 6; ++ks) {
#pragma unroll
            for (int j = 0; j < 8; ++j) {
                bf16x2 av = *(const bf16x2*)(arow + 32 * ks + a_k(j, half));
                afrag[ks][2 * j] = av.x; afrag[ks][2 * j + 1] = av.y;
            }
        }
    }

    for (int kt = wave; kt < NTIL; kt += 8) {
        int k0 = kt * 16;
        const __bf16* brow = EdB + (size_t)(k0 + ln15) * KP;
        v8f acc = {};
#pragma unroll
        for (int ks = 0; ks < 6; ++ks) {
            v16bf bb;
#pragma unroll
            for (int j = 0; j < 8; ++j) {
                bf16x2 bv = *(const bf16x2*)(brow + 32 * ks + a_k(j, half));
                bb[2 * j] = bv.x; bb[2 * j + 1] = bv.y;
            }
            acc = __builtin_amdgcn_wmma_f32_16x16x32_bf16(
                false, afrag[ks], false, bb, (short)0, acc, false, false);
        }
        const float scale = 0.08838834764831845f;  // 1/sqrt(128)
#pragma unroll
        for (int r = 0; r < 8; ++r) {
            float v = acc[r] * scale;
            sc[r + 8 * half][k0 + ln15] = v > 0.f ? v : 0.f;
        }
    }
    __syncthreads();

    // Softmax per row (wave32 shuffle reductions) + batch-mean accumulation
    for (int r = wave; r < 16; r += 8) {
        int n = n0 + r;
        if (n >= NN) continue;
        float mx = -1e30f;
        for (int k = lane; k < NN; k += 32) mx = fmaxf(mx, sc[r][k]);
#pragma unroll
        for (int o = 16; o > 0; o >>= 1) mx = fmaxf(mx, __shfl_xor(mx, o, 32));
        float sum = 0.f;
        for (int k = lane; k < NN; k += 32) sum += __expf(sc[r][k] - mx);
#pragma unroll
        for (int o = 16; o > 0; o >>= 1) sum += __shfl_xor(sum, o, 32);
        float inv = 1.f / (sum * (float)BATCH);
        for (int k = lane; k < NN; k += 32) {
            float p = __expf(sc[r][k] - mx) * inv;
            atomicAdd(&accum[n * NN + k], p);
        }
    }
}

// ---------------- Kernel 3: threshold mean > 0.5 -> {0,1} -------------------
__global__ void threshold_kernel(const float* __restrict__ accum, float* __restrict__ out) {
    int i = blockIdx.x * 256 + threadIdx.x;
    if (i < NN * NN) out[i] = accum[i] > 0.5f ? 1.f : 0.f;
}

extern "C" void kernel_launch(void* const* d_in, const int* in_sizes, int n_in,
                              void* d_out, int out_size, void* d_ws, size_t ws_size,
                              hipStream_t stream) {
    (void)in_sizes; (void)n_in; (void)out_size; (void)ws_size;
    const float* x  = (const float*)d_in[0];   // [B,C,N,T] f32
    const float* Es = (const float*)d_in[1];   // [C,N] f32
    float* out = (float*)d_out;                // [N,N] f32

    char* ws = (char*)d_ws;
    float*  accum = (float*)ws;                        // 170*170*4 = 115,600 B
    __bf16* EsT   = (__bf16*)(ws + 131072);            // 176*128*2 =  45,056 B
    __bf16* Ed    = (__bf16*)(ws + 131072 + 65536);    // 128*176*192*2 ~ 8.65 MB

    hipMemsetAsync(accum, 0, NN * NN * sizeof(float), stream);
    es_convert_kernel<<<(NP * CH + 255) / 256, 256, 0, stream>>>(Es, EsT);
    gemm1_tanh_kernel<<<BATCH * NTIL, 256, 0, stream>>>(x, EsT, Ed);
    gemm2_softmax_kernel<<<BATCH * NTIL, 256, 0, stream>>>(Ed, accum);
    threshold_kernel<<<(NN * NN + 255) / 256, 256, 0, stream>>>(accum, out);
}
